// GraphAttentionLayer_77197742178939
// MI455X (gfx1250) — matrix operand order repaired and checked by reading, not compiled, not run on hardware
//
#include <hip/hip_runtime.h>
#include <hip/hip_bf16.h>
#include <cmath>

typedef __attribute__((ext_vector_type(16))) _Float16 v16h;
typedef __attribute__((ext_vector_type(8)))  _Float16 v8h;
typedef __attribute__((ext_vector_type(8)))  float    v8f;
typedef __attribute__((ext_vector_type(2)))  float    v2f;

#define GAT_NEG   (-9000000000000000.0f)
#define GAT_ALPHA (0.2f)

constexpr int GN    = 8192;
constexpr int GFIN  = 512;
constexpr int GFOUT = 64;
constexpr int SPLIT = 8;    // waves per row-block (flash split over columns)
constexpr int TILE  = 32;   // K-tile (columns per WMMA step)

// ---------------------------------------------------------------------------
// Kernel A: wh = x @ w  (f32 WMMA 16x16x4), also emit wh_t (f16, 64 x 8192)
// One wave per 16 rows; 4 waves per block.
// f32 A layout: lanes0-15 {K=k,k+1}, lanes16-31 {K=k+2,k+3}; B mirrors on K.
// ---------------------------------------------------------------------------
__global__ __launch_bounds__(128)
void gat_wh_kernel(const float* __restrict__ x, const float* __restrict__ w,
                   float* __restrict__ wh, _Float16* __restrict__ wht) {
  const int lane = threadIdx.x & 31;
  const int wv   = threadIdx.x >> 5;
  const int half = lane >> 4;
  const int lr   = lane & 15;
  const int row0 = blockIdx.x * 64 + wv * 16;
  const int row  = row0 + lr;

  v8f acc0 = {}, acc1 = {}, acc2 = {}, acc3 = {};
  const float* xrow = x + (size_t)row * GFIN;

  for (int k = 0; k < GFIN; k += 4) {
    const int ka = k + 2 * half;
    v2f a;
    a.x = xrow[ka];
    a.y = xrow[ka + 1];
    const float* wp = w + (size_t)ka * GFOUT + lr;
    v2f b0, b1, b2, b3;
    b0.x = wp[0];  b0.y = wp[GFOUT + 0];
    b1.x = wp[16]; b1.y = wp[GFOUT + 16];
    b2.x = wp[32]; b2.y = wp[GFOUT + 32];
    b3.x = wp[48]; b3.y = wp[GFOUT + 48];
    acc0 = __builtin_amdgcn_wmma_f32_16x16x4_f32(false, a, false, b0, (short)0, acc0, false, false);
    acc1 = __builtin_amdgcn_wmma_f32_16x16x4_f32(false, a, false, b1, (short)0, acc1, false, false);
    acc2 = __builtin_amdgcn_wmma_f32_16x16x4_f32(false, a, false, b2, (short)0, acc2, false, false);
    acc3 = __builtin_amdgcn_wmma_f32_16x16x4_f32(false, a, false, b3, (short)0, acc3, false, false);
  }

  // C layout: lane n (lanes16-31 -> +8 rows); VGPR r -> row r; col = chunk*16 + lr
  float* whp = wh + (size_t)(row0 + 8 * half) * GFOUT + lr;
#pragma unroll
  for (int r = 0; r < 8; ++r) {
    whp[(size_t)r * GFOUT + 0]  = acc0[r];
    whp[(size_t)r * GFOUT + 16] = acc1[r];
    whp[(size_t)r * GFOUT + 32] = acc2[r];
    whp[(size_t)r * GFOUT + 48] = acc3[r];
  }
  // wh_t[col][row] f16: per chunk, 8 consecutive rows -> one 16B store
  {
    v8h h0, h1, h2, h3;
#pragma unroll
    for (int r = 0; r < 8; ++r) {
      h0[r] = (_Float16)acc0[r];
      h1[r] = (_Float16)acc1[r];
      h2[r] = (_Float16)acc2[r];
      h3[r] = (_Float16)acc3[r];
    }
    const int rbase = row0 + 8 * half;
    *(v8h*)(wht + (size_t)(0  + lr) * GN + rbase) = h0;
    *(v8h*)(wht + (size_t)(16 + lr) * GN + rbase) = h1;
    *(v8h*)(wht + (size_t)(32 + lr) * GN + rbase) = h2;
    *(v8h*)(wht + (size_t)(48 + lr) * GN + rbase) = h3;
  }
}

// ---------------------------------------------------------------------------
// Kernel B: s1 = wh @ a[:64], s2 = wh @ a[64:]
// ---------------------------------------------------------------------------
__global__ __launch_bounds__(256)
void gat_s_kernel(const float* __restrict__ wh, const float* __restrict__ a,
                  float* __restrict__ s1, float* __restrict__ s2) {
  const int r = blockIdx.x * blockDim.x + threadIdx.x;
  if (r >= GN) return;
  const float* p = wh + (size_t)r * GFOUT;
  float t1 = 0.f, t2 = 0.f;
#pragma unroll 8
  for (int i = 0; i < GFOUT; ++i) {
    const float v = p[i];
    t1 += v * a[i];
    t2 += v * a[GFOUT + i];
  }
  s1[r] = t1;
  s2[r] = t2;
}

// ---------------------------------------------------------------------------
// Kernel C: fused masked-softmax attention + P@wh (flash style, split-K)
// Block = 256 threads = 8 waves, all on the same 16-row block; wave wv takes
// column tiles t = wv, wv+8, ... Partial (m, l, acc) merged through LDS.
// ---------------------------------------------------------------------------
__global__ __launch_bounds__(256)
void gat_attn_kernel(const int* __restrict__ adj, const float* __restrict__ s1,
                     const float* __restrict__ s2, const _Float16* __restrict__ wht,
                     float* __restrict__ out) {
  __shared__ float lds_acc[SPLIT][16][GFOUT];
  __shared__ float lds_m[SPLIT][16];
  __shared__ float lds_l[SPLIT][16];

  const int lane = threadIdx.x & 31;
  const int wv   = threadIdx.x >> 5;
  const int half = lane >> 4;
  const int lr   = lane & 15;
  const int row0 = blockIdx.x * 16;
  const int row  = row0 + lr;

  const float s1v = s1[row];
  v8f acc0 = {}, acc1 = {}, acc2 = {}, acc3 = {};
  float m = GAT_NEG;
  float l = 0.f;

  const int* adjrow = adj + (size_t)row * GN + half * 8;

  for (int t = wv; t < GN / TILE; t += SPLIT) {
    const int jj = t * TILE;
    // adj + s2 for this lane's 16 columns: jj+half*8+[0..7], jj+16+half*8+[0..7]
    const int* ap = adjrow + jj;
    const int4 A0 = *(const int4*)(ap);
    const int4 A1 = *(const int4*)(ap + 4);
    const int4 A2 = *(const int4*)(ap + 16);
    const int4 A3 = *(const int4*)(ap + 20);
    const float* sp = s2 + jj + half * 8;
    const float4 Q0 = *(const float4*)(sp);
    const float4 Q1 = *(const float4*)(sp + 4);
    const float4 Q2 = *(const float4*)(sp + 16);
    const float4 Q3 = *(const float4*)(sp + 20);

    auto mke = [&](int av, float s2v) {
      float e = s1v + s2v;
      e = e > 0.f ? e : GAT_ALPHA * e;
      return av > 0 ? e : GAT_NEG;
    };
    float ev[16];
    ev[0]  = mke(A0.x, Q0.x); ev[1]  = mke(A0.y, Q0.y);
    ev[2]  = mke(A0.z, Q0.z); ev[3]  = mke(A0.w, Q0.w);
    ev[4]  = mke(A1.x, Q1.x); ev[5]  = mke(A1.y, Q1.y);
    ev[6]  = mke(A1.z, Q1.z); ev[7]  = mke(A1.w, Q1.w);
    ev[8]  = mke(A2.x, Q2.x); ev[9]  = mke(A2.y, Q2.y);
    ev[10] = mke(A2.z, Q2.z); ev[11] = mke(A2.w, Q2.w);
    ev[12] = mke(A3.x, Q3.x); ev[13] = mke(A3.y, Q3.y);
    ev[14] = mke(A3.z, Q3.z); ev[15] = mke(A3.w, Q3.w);

    // online-softmax: row max across this tile (lane pair L,L+16 share a row)
    float tm = ev[0];
#pragma unroll
    for (int i = 1; i < 16; ++i) tm = fmaxf(tm, ev[i]);
    tm = fmaxf(tm, __shfl_xor(tm, 16, 32));
    const float mnew  = fmaxf(m, tm);
    const float scale = __expf(m - mnew);
    m = mnew;

    float ps = 0.f;
    v16h pa;
#pragma unroll
    for (int i = 0; i < 16; ++i) {
      const float p = __expf(ev[i] - mnew);
      ps += p;
      pa[i] = (_Float16)p;       // P in f16 A-layout for WMMA
    }
    l = l * scale + ps;

    // rescale accumulators: acc VGPR r holds row (8*half + r) -> broadcast scale
#pragma unroll
    for (int r = 0; r < 8; ++r) {
      const float sr = __shfl(scale, (half << 3) + r, 32);
      acc0[r] *= sr; acc1[r] *= sr; acc2[r] *= sr; acc3[r] *= sr;
    }

    // B tiles (f16, 32x16 each): lane n holds K = half*8 + [0..7], 16 + half*8 + [0..7] of col n
    const _Float16* bp = wht + jj + half * 8;
    auto ldb = [&](const _Float16* q) {
      const v8h lo = *(const v8h*)(q);
      const v8h hi = *(const v8h*)(q + 16);
      return (v16h)__builtin_shufflevector(lo, hi, 0,1,2,3,4,5,6,7,8,9,10,11,12,13,14,15);
    };
    const v16h b0 = ldb(bp + (size_t)(0  + lr) * GN);
    const v16h b1 = ldb(bp + (size_t)(16 + lr) * GN);
    const v16h b2 = ldb(bp + (size_t)(32 + lr) * GN);
    const v16h b3 = ldb(bp + (size_t)(48 + lr) * GN);

    acc0 = __builtin_amdgcn_wmma_f32_16x16x32_f16(false, pa, false, b0, (short)0, acc0, false, false);
    acc1 = __builtin_amdgcn_wmma_f32_16x16x32_f16(false, pa, false, b1, (short)0, acc1, false, false);
    acc2 = __builtin_amdgcn_wmma_f32_16x16x32_f16(false, pa, false, b2, (short)0, acc2, false, false);
    acc3 = __builtin_amdgcn_wmma_f32_16x16x32_f16(false, pa, false, b3, (short)0, acc3, false, false);
  }

  // merge split-K partials through LDS
  const float lt = l + __shfl_xor(l, 16, 32);
  if (half == 0) { lds_m[wv][lr] = m; lds_l[wv][lr] = lt; }
#pragma unroll
  for (int r = 0; r < 8; ++r) {
    lds_acc[wv][8 * half + r][0  + lr] = acc0[r];
    lds_acc[wv][8 * half + r][16 + lr] = acc1[r];
    lds_acc[wv][8 * half + r][32 + lr] = acc2[r];
    lds_acc[wv][8 * half + r][48 + lr] = acc3[r];
  }
  __syncthreads();

  const int c  = threadIdx.x & 63;
  const int rb = threadIdx.x >> 6;
#pragma unroll
  for (int q = 0; q < 4; ++q) {
    const int r = rb + 4 * q;
    float mst = lds_m[0][r];
#pragma unroll
    for (int wq = 1; wq < SPLIT; ++wq) mst = fmaxf(mst, lds_m[wq][r]);
    float den = 0.f, num = 0.f;
#pragma unroll
    for (int wq = 0; wq < SPLIT; ++wq) {
      const float f = __expf(lds_m[wq][r] - mst);
      den += f * lds_l[wq][r];
      num += f * lds_acc[wq][r][c];
    }
    const float h = num / den;
    out[(size_t)(row0 + r) * GFOUT + c] = h > 0.f ? h : expm1f(h);
  }
}

// ---------------------------------------------------------------------------
extern "C" void kernel_launch(void* const* d_in, const int* in_sizes, int n_in,
                              void* d_out, int out_size, void* d_ws, size_t ws_size,
                              hipStream_t stream) {
  const float* x   = (const float*)d_in[0];
  const int*   adj = (const int*)d_in[1];
  const float* w   = (const float*)d_in[2];
  const float* a   = (const float*)d_in[3];
  float* out = (float*)d_out;

  char* ws = (char*)d_ws;
  float*    wh  = (float*)ws;                                   // 8192*64 f32   (2 MB)
  _Float16* wht = (_Float16*)(ws + (size_t)GN * GFOUT * 4);     // 64*8192 f16   (1 MB)
  float*    s1  = (float*)(ws + (size_t)GN * GFOUT * 6);        // 8192 f32
  float*    s2  = s1 + GN;                                      // 8192 f32

  gat_wh_kernel<<<GN / 64, 128, 0, stream>>>(x, w, wh, wht);
  gat_s_kernel<<<GN / 256, 256, 0, stream>>>(wh, a, s1, s2);
  gat_attn_kernel<<<GN / 16, 256, 0, stream>>>(adj, s1, s2, wht, out);
}
